// GlobalUpdate_3685081940032
// MI455X (gfx1250) — matrix operand order repaired and checked by reading, not compiled, not run on hardware
//
#include <hip/hip_runtime.h>

// ---- problem constants (from reference) ------------------------------------
#define N_ROWS 4096
#define D_IN   256
#define H_HEADS 8
#define DFF    512            // D * FACTOR
#define DH     64             // DFF / H
#define NCHAIN 64
#define NBATCH 8
#define DB     1024           // 2 * DFF

typedef __attribute__((ext_vector_type(16))) _Float16 v16h;
typedef __attribute__((ext_vector_type(8)))  _Float16 v8h;
typedef __attribute__((ext_vector_type(8)))  float    v8f;

__device__ __forceinline__ float gelu_tanh(float x) {
    // jax.nn.gelu default (approximate=True, tanh form)
    float x3 = x * x * x;
    return 0.5f * x * (1.0f + tanhf(0.7978845608028654f * (x + 0.044715f * x3)));
}

// ---- f32 -> f16 elementwise -------------------------------------------------
__global__ void conv_f16_kernel(const float* __restrict__ src,
                                _Float16* __restrict__ dst, int n) {
    int i = blockIdx.x * blockDim.x + threadIdx.x;
    if (i < n) dst[i] = (_Float16)src[i];
}

// ---- f32 [Krows][Ncols] -> f16 transposed [Ncols][Krows] -------------------
__global__ void transpose_conv_kernel(const float* __restrict__ src,
                                      _Float16* __restrict__ dst,
                                      int Krows, int Ncols) {
    int i = blockIdx.x * blockDim.x + threadIdx.x;
    int total = Krows * Ncols;
    if (i >= total) return;
    int n = i / Krows;
    int k = i - n * Krows;
    dst[(size_t)n * Krows + k] = (_Float16)src[(size_t)k * Ncols + n];
}

// ---- WMMA GEMM: C[M][N] = A[M][K] * BT[N][K]^T (+bias, optional gelu) ------
// Block: 256 threads = 8 waves; block tile 128(M) x 256(N); wave tile 64x64.
// K-chunks of 32 via v_wmma_f32_16x16x32_f16 (f16 in, f32 accumulate).
__global__ __launch_bounds__(256) void wmma_gemm_kernel(
    const _Float16* __restrict__ A,   // [M][K] row-major
    const _Float16* __restrict__ BT,  // [N][K] row-major (pre-transposed B)
    const float*    __restrict__ bias, // [N] or nullptr
    float*          __restrict__ C,   // [M][N]
    int M, int Nn, int Kk, int do_gelu)
{
    const int lane = threadIdx.x & 31;
    const int w    = threadIdx.x >> 5;
    const int wm   = w & 1;
    const int wn   = w >> 1;
    const int m0   = blockIdx.y * 128 + wm * 64;
    const int n0   = blockIdx.x * 256 + wn * 64;

    const v8f vzero = {0.f, 0.f, 0.f, 0.f, 0.f, 0.f, 0.f, 0.f};
    v8f acc[4][4];
#pragma unroll
    for (int i = 0; i < 4; ++i)
#pragma unroll
        for (int j = 0; j < 4; ++j) acc[i][j] = vzero;

    // Per-lane fragment addressing (CDNA5 16-bit A 16x32 / B 32x16 layouts):
    // A: lane L holds M=(L&15); halves 0..7 = K kb..kb+7, halves 8..15 = K kb+16..kb+23, kb=(L>=16)*8
    // B: lane L holds N=(L&15); halves 0..7 = K kb2..kb2+7, halves 8..15 = K kb2+16..kb2+23, kb2=(L>=16)*8
    const int mrow = m0 + (lane & 15);
    const int nrow = n0 + (lane & 15);
    const int kbA  = (lane >> 4) * 8;
    const int kbB  = (lane >> 4) * 8;

    for (int kc = 0; kc < Kk; kc += 32) {
        if (kc + 32 < Kk) {  // emits global_prefetch_b8 for next K-chunk
            __builtin_prefetch(A + (size_t)mrow * Kk + kc + 32 + kbA, 0, 0);
            __builtin_prefetch(BT + (size_t)nrow * Kk + kc + 32 + kbB, 0, 0);
        }
        v16h af[4], bf[4];
#pragma unroll
        for (int i = 0; i < 4; ++i) {
            const _Float16* pa = A + (size_t)(mrow + i * 16) * Kk + kc + kbA;
            v8h lo = *(const v8h*)pa;
            v8h hi = *(const v8h*)(pa + 16);
#pragma unroll
            for (int j = 0; j < 8; ++j) { af[i][j] = lo[j]; af[i][j + 8] = hi[j]; }
        }
#pragma unroll
        for (int j = 0; j < 4; ++j) {
            const _Float16* pb = BT + (size_t)(nrow + j * 16) * Kk + kc + kbB;
            v8h lo = *(const v8h*)pb;
            v8h hi = *(const v8h*)(pb + 16);
#pragma unroll
            for (int q = 0; q < 8; ++q) { bf[j][q] = lo[q]; bf[j][q + 8] = hi[q]; }
        }
#pragma unroll
        for (int i = 0; i < 4; ++i)
#pragma unroll
            for (int j = 0; j < 4; ++j)
                acc[i][j] = __builtin_amdgcn_wmma_f32_16x16x32_f16(
                    false, af[i], false, bf[j], (short)0, acc[i][j], false, false);
    }

    // Epilogue. f32 C/D layout: VGPR r: lanes 0-15 -> M=r, lanes 16-31 -> M=8+r; N=(L&15)
    const int rbase = (lane >> 4) * 8;
#pragma unroll
    for (int i = 0; i < 4; ++i) {
#pragma unroll
        for (int j = 0; j < 4; ++j) {
            const int col = n0 + j * 16 + (lane & 15);
            const float bv = bias ? bias[col] : 0.f;
#pragma unroll
            for (int r = 0; r < 8; ++r) {
                const int row = m0 + i * 16 + rbase + r;
                float v = acc[i][j][r] + bv;
                if (do_gelu) v = gelu_tanh(v);
                C[(size_t)row * Nn + col] = v;
            }
        }
    }
}

// ---- build deterministic per-segment index lists (masked rows only) --------
__global__ void build_lists_kernel(const int* __restrict__ chain,
                                   const int* __restrict__ batch,
                                   const unsigned char* __restrict__ mask,
                                   int* __restrict__ listC, int* __restrict__ cntC,
                                   int* __restrict__ listB, int* __restrict__ cntB)
{
    int t = threadIdx.x;
    if (t < NCHAIN) {
        int n = 0;
        for (int i = 0; i < N_ROWS; ++i)
            if (chain[i] == t && mask[i]) listC[t * N_ROWS + (n++)] = i;
        cntC[t] = n;
    } else if (t < NCHAIN + NBATCH) {
        int b = t - NCHAIN;
        int n = 0;
        for (int i = 0; i < N_ROWS; ++i)
            if (batch[i] == b && mask[i]) listB[b * N_ROWS + (n++)] = i;
        cntB[b] = n;
    }
}

// ---- per-(segment, head) masked outer-product mean: S = K^T V / count ------
// grid.x = nseg*8; block = 256 threads; thread owns row a=t>>2, 16 cols b0..b0+15
__global__ __launch_bounds__(256) void outer_accum_kernel(
    const float* __restrict__ Kf,   // [N_ROWS][DFF] (gelu'd key proj)
    const float* __restrict__ Vf,   // [N_ROWS][DFF]
    const int*   __restrict__ lists, const int* __restrict__ cnts,
    float* __restrict__ S)          // [nseg][H][64][64] (already mean-scaled)
{
    const int seg = blockIdx.x >> 3;
    const int h   = blockIdx.x & 7;
    const int t   = threadIdx.x;
    const int a   = t >> 2;
    const int b0  = (t & 3) * 16;
    const int n   = cnts[seg];
    const int* lst = lists + (size_t)seg * N_ROWS;

    float acc[16];
#pragma unroll
    for (int j = 0; j < 16; ++j) acc[j] = 0.f;

    for (int r = 0; r < n; ++r) {
        const int idx = lst[r];
        const float  kv = Kf[(size_t)idx * DFF + h * DH + a];
        const float4* v4 = (const float4*)(Vf + (size_t)idx * DFF + h * DH + b0);
        float4 v0 = v4[0], v1 = v4[1], v2 = v4[2], v3 = v4[3];
        acc[0]  += kv * v0.x; acc[1]  += kv * v0.y; acc[2]  += kv * v0.z; acc[3]  += kv * v0.w;
        acc[4]  += kv * v1.x; acc[5]  += kv * v1.y; acc[6]  += kv * v1.z; acc[7]  += kv * v1.w;
        acc[8]  += kv * v2.x; acc[9]  += kv * v2.y; acc[10] += kv * v2.z; acc[11] += kv * v2.w;
        acc[12] += kv * v3.x; acc[13] += kv * v3.y; acc[14] += kv * v3.z; acc[15] += kv * v3.w;
    }
    const float scale = 1.f / fmaxf((float)n, 1e-6f);
    float* so = S + ((((size_t)seg * H_HEADS + h) * 64 + a) * 64 + b0);
#pragma unroll
    for (int j = 0; j < 16; ++j) so[j] = acc[j] * scale;
}

// ---- per-batch mean of bias projection -------------------------------------
__global__ __launch_bounds__(256) void bias_mean_kernel(
    const float* __restrict__ Bp,    // [N_ROWS][DB]
    const int* __restrict__ listB, const int* __restrict__ cntB,
    float* __restrict__ biasmean)    // [NBATCH][DB]
{
    const int b  = blockIdx.x;
    const int c0 = threadIdx.x * 4;
    const int n  = cntB[b];
    const int* lst = listB + (size_t)b * N_ROWS;
    float a0 = 0.f, a1 = 0.f, a2 = 0.f, a3 = 0.f;
    for (int r = 0; r < n; ++r) {
        const int idx = lst[r];
        float4 x = *(const float4*)(Bp + (size_t)idx * DB + c0);
        a0 += x.x; a1 += x.y; a2 += x.z; a3 += x.w;
    }
    const float scale = 1.f / fmaxf((float)n, 1e-6f);
    float* o = biasmean + (size_t)b * DB + c0;
    o[0] = a0 * scale; o[1] = a1 * scale; o[2] = a2 * scale; o[3] = a3 * scale;
}

// ---- gathered einsum + bias add, f16 output for final WMMA GEMM ------------
// block per row i; LDS-stage q[i, 0:512]; thread computes 4 output columns
__global__ __launch_bounds__(256) void einsum_kernel(
    const float* __restrict__ Qf,       // [N_ROWS][DFF] (gelu'd query proj)
    const float* __restrict__ Schain,   // [NCHAIN][H][64][64]
    const float* __restrict__ Sbatch,   // [NBATCH][H][64][64]
    const float* __restrict__ biasmean, // [NBATCH][DB]
    const int* __restrict__ chain, const int* __restrict__ batch,
    _Float16* __restrict__ E)           // [N_ROWS][DB]
{
    __shared__ float qs[DFF];
    const int i = blockIdx.x;
    const int t = threadIdx.x;
    qs[t]       = Qf[(size_t)i * DFF + t];
    qs[t + 256] = Qf[(size_t)i * DFF + t + 256];
    __syncthreads();

    const int c  = chain[i];
    const int bb = batch[i];
#pragma unroll
    for (int u = 0; u < 4; ++u) {
        const int col = t * 4 + u;
        const int h   = col >> 7;       // 128 cols per head
        const int vv  = col & 127;      // [0,64): chain part, [64,128): batch part
        const float* srow = (vv < 64)
            ? Schain + (((size_t)c  * H_HEADS + h) * 64 + vv) * 64
            : Sbatch + (((size_t)bb * H_HEADS + h) * 64 + (vv - 64)) * 64;
        const float* qh = qs + h * DH;
        float acc = 0.f;
#pragma unroll
        for (int kk = 0; kk < 64; ++kk) acc += srow[kk] * qh[kk];
        acc += biasmean[(size_t)bb * DB + col];
        E[(size_t)i * DB + col] = (_Float16)acc;
    }
}

// ---- host-side launcher -----------------------------------------------------
static inline size_t align_up(size_t x) { return (x + 255) & ~(size_t)255; }

extern "C" void kernel_launch(void* const* d_in, const int* in_sizes, int n_in,
                              void* d_out, int out_size, void* d_ws, size_t ws_size,
                              hipStream_t stream) {
    (void)in_sizes; (void)n_in; (void)out_size; (void)ws_size;

    const float* local   = (const float*)d_in[0];
    const int*   chain   = (const int*)d_in[1];
    const int*   batch   = (const int*)d_in[2];
    const unsigned char* mask = (const unsigned char*)d_in[3];
    const float* W_key   = (const float*)d_in[4];
    const float* b_key   = (const float*)d_in[5];
    const float* W_value = (const float*)d_in[6];
    const float* b_value = (const float*)d_in[7];
    const float* W_query = (const float*)d_in[8];
    const float* b_query = (const float*)d_in[9];
    const float* W_bias  = (const float*)d_in[10];
    const float* b_bias  = (const float*)d_in[11];
    const float* W_out   = (const float*)d_in[12];
    float* out = (float*)d_out;

    // workspace carve-out
    char* ws = (char*)d_ws;
    size_t off = 0;
    auto take = [&](size_t bytes) { char* p = ws + off; off = align_up(off + bytes); return p; };

    _Float16* localh = (_Float16*)take((size_t)N_ROWS * D_IN * 2);
    _Float16* WkT    = (_Float16*)take((size_t)DFF * D_IN * 2);
    _Float16* WvT    = (_Float16*)take((size_t)DFF * D_IN * 2);
    _Float16* WqT    = (_Float16*)take((size_t)DFF * D_IN * 2);
    _Float16* WbT    = (_Float16*)take((size_t)DB * D_IN * 2);
    _Float16* WoT    = (_Float16*)take((size_t)D_IN * DB * 2);   // [256][1024]
    float* Kf        = (float*)take((size_t)N_ROWS * DFF * 4);
    float* Vf        = (float*)take((size_t)N_ROWS * DFF * 4);
    float* Qf        = (float*)take((size_t)N_ROWS * DFF * 4);
    float* Bp        = (float*)take((size_t)N_ROWS * DB * 4);
    float* Schain    = (float*)take((size_t)NCHAIN * H_HEADS * 64 * 64 * 4);
    float* Sbatch    = (float*)take((size_t)NBATCH * H_HEADS * 64 * 64 * 4);
    float* biasmean  = (float*)take((size_t)NBATCH * DB * 4);
    int* listC       = (int*)take((size_t)NCHAIN * N_ROWS * 4);
    int* listB       = (int*)take((size_t)NBATCH * N_ROWS * 4);
    int* cntC        = (int*)take((size_t)NCHAIN * 4);
    int* cntB        = (int*)take((size_t)NBATCH * 4);
    _Float16* E      = (_Float16*)take((size_t)N_ROWS * DB * 2);

    // 1) precision conversion + weight transposes (B laid out [N][K] for WMMA)
    {
        int n = N_ROWS * D_IN;
        conv_f16_kernel<<<(n + 255) / 256, 256, 0, stream>>>(local, localh, n);
    }
    transpose_conv_kernel<<<(D_IN * DFF + 255) / 256, 256, 0, stream>>>(W_key,   WkT, D_IN, DFF);
    transpose_conv_kernel<<<(D_IN * DFF + 255) / 256, 256, 0, stream>>>(W_value, WvT, D_IN, DFF);
    transpose_conv_kernel<<<(D_IN * DFF + 255) / 256, 256, 0, stream>>>(W_query, WqT, D_IN, DFF);
    transpose_conv_kernel<<<(D_IN * DB  + 255) / 256, 256, 0, stream>>>(W_bias,  WbT, D_IN, DB);
    transpose_conv_kernel<<<(DB * D_IN  + 255) / 256, 256, 0, stream>>>(W_out,   WoT, DB, D_IN);

    // 2) deterministic segment index lists (masked rows only)
    build_lists_kernel<<<1, 128, 0, stream>>>(chain, batch, mask, listC, cntC, listB, cntB);

    // 3) projection GEMMs via WMMA (f16 in, f32 accumulate)
    wmma_gemm_kernel<<<dim3(DFF / 256, N_ROWS / 128), 256, 0, stream>>>(
        localh, WkT, b_key,   Kf, N_ROWS, DFF, D_IN, /*gelu=*/1);
    wmma_gemm_kernel<<<dim3(DFF / 256, N_ROWS / 128), 256, 0, stream>>>(
        localh, WvT, b_value, Vf, N_ROWS, DFF, D_IN, /*gelu=*/0);
    wmma_gemm_kernel<<<dim3(DFF / 256, N_ROWS / 128), 256, 0, stream>>>(
        localh, WqT, b_query, Qf, N_ROWS, DFF, D_IN, /*gelu=*/1);
    wmma_gemm_kernel<<<dim3(DB / 256, N_ROWS / 128), 256, 0, stream>>>(
        localh, WbT, b_bias,  Bp, N_ROWS, DB, D_IN, /*gelu=*/0);

    // 4) segment outer-product means (chain: 64 segs, batch: 8 segs)
    outer_accum_kernel<<<NCHAIN * H_HEADS, 256, 0, stream>>>(Kf, Vf, listC, cntC, Schain);
    outer_accum_kernel<<<NBATCH * H_HEADS, 256, 0, stream>>>(Kf, Vf, listB, cntB, Sbatch);

    // 5) per-batch bias mean
    bias_mean_kernel<<<NBATCH, 256, 0, stream>>>(Bp, listB, cntB, biasmean);

    // 6) gathered einsum + bias add -> E (f16)
    einsum_kernel<<<N_ROWS, 256, 0, stream>>>(Qf, Schain, Sbatch, biasmean, chain, batch, E);

    // 7) final GEMM: out = E @ W_out  (WMMA)
    wmma_gemm_kernel<<<dim3(D_IN / 256, N_ROWS / 128), 256, 0, stream>>>(
        E, WoT, nullptr, out, N_ROWS, D_IN, DB, /*gelu=*/0);
}
